// WaterNet0313_88794153878206
// MI455X (gfx1250) — matrix operand order repaired and checked by reading, not compiled, not run on hardware
//
#include <hip/hip_runtime.h>

// ---------------------------------------------------------------------------
// WaterNet fused persistent kernel for MI455X (gfx1250, wave32, WMMA).
// One workgroup (128 threads = 4 wave32) per site s (NS=2048 blocks).
// Dominant GEMM  pK = tanh(f@W_kin + b_kin + state) @ W_kout  is mapped to
// v_wmma_f32_16x16x32_f16:  M = 16 timesteps, N = 48 (3 tiles), K = 256.
// B (W_kout) fragments are loop-invariant and held in registers; only the
// A (activations) fragments stream from LDS, matching WGP LDS bandwidth.
// ---------------------------------------------------------------------------

typedef __attribute__((ext_vector_type(16))) _Float16 v16h;
typedef __attribute__((ext_vector_type(8)))  _Float16 v8h;
typedef __attribute__((ext_vector_type(2)))  _Float16 v2h;
typedef __attribute__((ext_vector_type(8)))  float    v8f;

#define NT 1000
#define NS 2048
#define NH 16
#define NR 8
#define HS 256
#define NG 32
#define TOUT (NT - NR + 1)            // 993
#define CH 16                         // timesteps per chunk (WMMA M)
#define NCHUNK ((NT + CH - 1) / CH)   // 63
#define HPAD 264                      // f16 row stride for h tile (16B-aligned rows)
#define NPK 48                        // pK width = NH*NK
#define NGATE 144                     // NH*9

__device__ __forceinline__ float sigmoidf_(float v) {
  return 1.0f / (1.0f + __expf(-v));
}

__global__ __launch_bounds__(128) void waternet_fused_kernel(
    const float* __restrict__ x,     // (NT,NS,6)
    const float* __restrict__ xc,    // (NS,NG)
    const float* __restrict__ Wfc,   // (NG,HS)
    const float* __restrict__ bfc,   // (HS)
    const float* __restrict__ Wr,    // (HS,144)
    const float* __restrict__ br,    // (144)
    const float* __restrict__ Wg,    // (HS,144)
    const float* __restrict__ bg,    // (144)
    const float* __restrict__ Wkin,  // (4,HS)
    const float* __restrict__ bkin,  // (HS)
    const float* __restrict__ Wkout, // (HS,48)
    const float* __restrict__ bkout, // (48)
    float* __restrict__ out)         // (TOUT,NS)
{
  __shared__ _Float16 h16[CH * HPAD];       // tanh activations, f16, 8.25 KB
  __shared__ float    pk[CH * NPK];         // WMMA result, 3 KB
  __shared__ float    stateL[HS];
  __shared__ float    hGL[HS];
  __shared__ float    xcL[NG];
  __shared__ float    pgL[NGATE];
  __shared__ float    prL[NGATE];
  __shared__ float    fL[CH * 4];           // T1,T2,Rad,Hum per chunk step
  __shared__ float    metL[CH * 3];         // Ps, Pl, Ev per chunk step
  __shared__ float    gateL[9 * NH];        // kp ks kd gd gl qb ga gi ge
  __shared__ float    wrL[NH * NR];

  const int s    = blockIdx.x;
  const int tid  = threadIdx.x;
  const int lane = tid & 31;                // wave32
  const int wid  = tid >> 5;

  // ------------------------- Phase 0: per-site statics ---------------------
  if (tid < NG) xcL[tid] = xc[s * NG + tid];
  __syncthreads();

  for (int o = tid; o < HS; o += 128) {
    float acc = bfc[o];
    #pragma unroll
    for (int j = 0; j < NG; ++j) acc += xcL[j] * Wfc[j * HS + o];
    stateL[o] = acc;
    hGL[o]    = tanhf(acc);
  }
  __syncthreads();

  for (int o = tid; o < NGATE; o += 128) {
    float ag = bg[o], ar = br[o];
    for (int j = 0; j < HS; ++j) {
      const float h = hGL[j];
      ag += h * Wg[j * NGATE + o];
      ar += h * Wr[j * NGATE + o];
    }
    pgL[o] = ag;
    prL[o] = ar;
  }
  __syncthreads();

  if (tid < NH) {
    const int h = tid;
    const float g0 = pgL[0 * NH + h], g1 = pgL[1 * NH + h], g2 = pgL[2 * NH + h];
    const float g3 = pgL[3 * NH + h], g4 = pgL[4 * NH + h], g5 = pgL[5 * NH + h];
    const float g6 = pgL[6 * NH + h], g7 = pgL[7 * NH + h], g8 = pgL[8 * NH + h];
    gateL[0 * NH + h] = sigmoidf_(g0);                       // kp
    gateL[1 * NH + h] = sigmoidf_(g1);                       // ks
    gateL[2 * NH + h] = sigmoidf_(g2);                       // kd
    gateL[3 * NH + h] = sigmoidf_(g3);                       // gd
    const float t5 = sigmoidf_(g4) * 10.0f;
    gateL[4 * NH + h] = t5 * t5 * t5;                        // gl
    gateL[5 * NH + h] = fmaxf(g5, 0.0f) * 0.1f;              // qb
    gateL[6 * NH + h] = g6;                                  // ga (raw, softmax later)
    gateL[7 * NH + h] = fminf(fmaxf(g7 * (1.0f / 6.0f) + 0.5f, 0.0f), 1.0f) * 0.5f; // gi
    gateL[8 * NH + h] = fmaxf(g8, 0.0f);                     // ge
    // routing weights wr[h][0..7]
    float rv[NR + 1];
    float mx = -1e30f;
    #pragma unroll
    for (int k = 0; k < NR + 1; ++k) { rv[k] = prL[h * (NR + 1) + k]; mx = fmaxf(mx, rv[k]); }
    float sum = 0.0f;
    #pragma unroll
    for (int k = 0; k < NR + 1; ++k) { rv[k] = __expf(rv[k] - mx); sum += rv[k]; }
    const float inv = 1.0f / sum;
    float c = 0.0f, wsum = 0.0f, w[NR];
    #pragma unroll
    for (int k = 0; k < NR; ++k) {
      const float sv = rv[k] * inv;
      c += sv;
      w[k] = fminf(c, 1.0f - c + sv);
      wsum += w[k];
    }
    const float invw = 1.0f / wsum;
    #pragma unroll
    for (int k = 0; k < NR; ++k) wrL[h * NR + k] = w[k] * invw;
  }
  __syncthreads();
  if (tid == 0) {  // softmax over h for ga (16 values, trivial)
    float mx = -1e30f;
    for (int h = 0; h < NH; ++h) mx = fmaxf(mx, gateL[6 * NH + h]);
    float sum = 0.0f, e[NH];
    for (int h = 0; h < NH; ++h) { e[h] = __expf(gateL[6 * NH + h] - mx); sum += e[h]; }
    const float inv = 1.0f / sum;
    for (int h = 0; h < NH; ++h) gateL[6 * NH + h] = e[h] * inv;
  }
  __syncthreads();

  // Per-thread W_kin columns (each thread owns hidden dims j0,j1)
  const int j0 = tid * 2, j1 = tid * 2 + 1;
  float wk0[4], wk1[4];
  #pragma unroll
  for (int c = 0; c < 4; ++c) { wk0[c] = Wkin[c * HS + j0]; wk1[c] = Wkin[c * HS + j1]; }
  const float bs0 = bkin[j0] + stateL[j0];
  const float bs1 = bkin[j1] + stateL[j1];

  // Loop-invariant B fragments (W_kout) in registers, waves 0..2 (N-tile = wid).
  // B 32x16 f16 layout: lane owns one N column; lane<16 -> K=kc+[0..7],kc+[16..23], else +8.
  v16h bfrag[HS / 32];
  {
    const int col   = (wid < 3 ? wid : 0) * 16 + (lane & 15);
    const int ksub  = (lane >> 4) * 8;
    #pragma unroll
    for (int kk = 0; kk < HS / 32; ++kk) {
      const int kbase = kk * 32 + ksub;
      #pragma unroll
      for (int i = 0; i < 16; ++i)
        bfrag[kk][i] = (_Float16)Wkout[(size_t)(kbase + (i < 8 ? i : i + 8)) * NPK + col];
    }
  }

  // Scan-lane registers (wave 0 lanes 0..15 own h-units; 16..31 mirror harmlessly)
  const int h = lane & (NH - 1);
  const float r_kp = gateL[0 * NH + h], r_ks = gateL[1 * NH + h];
  const float r_kd = gateL[2 * NH + h], r_gd = gateL[3 * NH + h];
  const float r_gl = gateL[4 * NH + h], r_qb = gateL[5 * NH + h];
  const float r_ga = gateL[6 * NH + h], r_gi = gateL[7 * NH + h];
  const float r_ge = gateL[8 * NH + h];
  float r_wr[NR];
  #pragma unroll
  for (int k = 0; k < NR; ++k) r_wr[k] = wrL[h * NR + k];
  const float bk0 = bkout[h], bk1 = bkout[NH + h], bk2 = bkout[2 * NH + h];
  float Sf = 0.0f, Ss = 0.0f, Sg = 0.0f;
  float accQ[NR];
  #pragma unroll
  for (int j = 0; j < NR; ++j) accQ[j] = 0.0f;

  // ------------------------- Main time loop (63 chunks of 16) --------------
  for (int ch = 0; ch < NCHUNK; ++ch) {
    const int t0 = ch * CH;
    __syncthreads();  // previous chunk's scan finished reading pk/metL

    if (tid < CH) {   // load this chunk's forcings
      const int t = t0 + tid;
      float prcp = 0.f, evp = 0.f, T1v = 0.f, T2v = 0.f, rad = 0.f, hum = 0.f;
      if (t < NT) {
        const float* xp = x + ((size_t)t * NS + s) * 6;
        prcp = xp[0]; evp = xp[1]; T1v = xp[2]; T2v = xp[3]; rad = xp[4]; hum = xp[5];
      }
      fL[tid * 4 + 0] = T1v; fL[tid * 4 + 1] = T2v;
      fL[tid * 4 + 2] = rad; fL[tid * 4 + 3] = hum;
      const float fl = fminf(fmaxf(T2v / (T2v - T1v + 1e-5f), 0.0f), 1.0f);
      metL[tid * 3 + 0] = prcp * (1.0f - fl);  // Ps
      metL[tid * 3 + 1] = prcp * fl;           // Pl
      metL[tid * 3 + 2] = evp;                 // Ev
    }
    __syncthreads();

    // Phase 1: h16[tt][j] = tanh(f@W_kin + b_kin + state), packed v2h stores
    #pragma unroll 4
    for (int tt = 0; tt < CH; ++tt) {
      const float f0 = fL[tt * 4 + 0], f1 = fL[tt * 4 + 1];
      const float f2 = fL[tt * 4 + 2], f3 = fL[tt * 4 + 3];
      const float a0 = bs0 + f0 * wk0[0] + f1 * wk0[1] + f2 * wk0[2] + f3 * wk0[3];
      const float a1 = bs1 + f0 * wk1[0] + f1 * wk1[1] + f2 * wk1[2] + f3 * wk1[3];
      v2h p;
      p[0] = (_Float16)tanhf(a0);
      p[1] = (_Float16)tanhf(a1);
      *(v2h*)&h16[tt * HPAD + j0] = p;
    }
    __syncthreads();

    // Phase 2: pK tile = h(16x256) @ Wkout(256x48), waves 0..2 own N-tiles
    if (wid < 3) {
      const int mrow = lane & 15;
      const int hi   = lane >> 4;     // lane half selects K sub-range per ISA layout
      v8f c = {};
      const _Float16* hrow = &h16[mrow * HPAD + hi * 8];
      #pragma unroll
      for (int kc = 0; kc < HS; kc += 32) {
        // A 16x32 f16: lane<16 -> K = kc+[0..7], kc+[16..23]; lane>=16 -> +8 shift
        const v8h a0 = *(const v8h*)(hrow + kc);
        const v8h a1 = *(const v8h*)(hrow + kc + 16);
        const v16h a = __builtin_shufflevector(a0, a1,
            0, 1, 2, 3, 4, 5, 6, 7, 8, 9, 10, 11, 12, 13, 14, 15);
        c = __builtin_amdgcn_wmma_f32_16x16x32_f16(
            false, a, false, bfrag[kc / 32], (short)0, c, false, false);
      }
      // C layout: VGPR r -> M=r (lane<16) / M=8+r (lane>=16); N = lane&15
      const int nc = wid * 16 + (lane & 15);
      const int mb = hi * 8;
      #pragma unroll
      for (int r = 0; r < 8; ++r) pk[(mb + r) * NPK + nc] = c[r];
    }
    __syncthreads();

    // Phase 3: sequential bucket scan + FIR routing (wave 0, lanes = h-units)
    if (wid == 0) {
      for (int tt = 0; tt < CH; ++tt) {
        const int t = t0 + tt;
        if (t >= NT) break;
        const float km = __expf(pk[tt * NPK + h] + bk0);
        const float ki = fmaxf(pk[tt * NPK + NH + h] + bk1, 0.0f);
        const float ke = __expf(pk[tt * NPK + 2 * NH + h] + bk2);
        const float Ps_t = metL[tt * 3 + 0];
        const float Pl_t = metL[tt * 3 + 1];
        const float Ev_t = metL[tt * 3 + 2];

        Sf += Ps_t;
        const float qf = fminf(Sf, km);
        Sf -= qf;
        const float inflow = qf + Pl_t;
        const float qd = inflow * r_gi;
        Ss += inflow * (1.0f - r_gi);
        const float E = fminf(fmaxf(Ss, 0.0f), Ev_t * ke * r_ge);
        Ss -= E;
        const float qi = fminf(ki, fmaxf(Ss, 0.0f));
        Ss -= qi;
        const float qp  = r_kp * fmaxf(Ss - r_gl, 0.0f);
        const float qsa = r_ks * fminf(fmaxf(Ss, 0.0f), r_gl);
        Ss -= qp + qsa;
        Sg += qsa * r_gd;
        const float qg = r_kd * Sg + r_qb;
        Sg *= (1.0f - r_kd);
        const float q = qp + qsa * (1.0f - r_gd) + qg + qd + qi;

        // 8-tap routing ring: accQ[j] is output (t-j); weight wr[7-j]
        #pragma unroll
        for (int j = 0; j < NR; ++j) accQ[j] += q * r_wr[NR - 1 - j];
        if (t >= NR - 1) {
          float v = accQ[NR - 1] * r_ga;  // weighted by ga, reduce over 16 h-lanes
          v += __shfl_xor(v, 1, 32);
          v += __shfl_xor(v, 2, 32);
          v += __shfl_xor(v, 4, 32);
          v += __shfl_xor(v, 8, 32);
          if (lane == 0) out[(size_t)(t - (NR - 1)) * NS + s] = v;
        }
        #pragma unroll
        for (int j = NR - 1; j > 0; --j) accQ[j] = accQ[j - 1];
        accQ[0] = 0.0f;
      }
    }
  }
}

extern "C" void kernel_launch(void* const* d_in, const int* in_sizes, int n_in,
                              void* d_out, int out_size, void* d_ws, size_t ws_size,
                              hipStream_t stream) {
  (void)in_sizes; (void)n_in; (void)out_size; (void)d_ws; (void)ws_size;
  waternet_fused_kernel<<<NS, 128, 0, stream>>>(
      (const float*)d_in[0],   // x
      (const float*)d_in[1],   // xc
      (const float*)d_in[2],   // W_fc
      (const float*)d_in[3],   // b_fc
      (const float*)d_in[4],   // W_r
      (const float*)d_in[5],   // b_r
      (const float*)d_in[6],   // W_g
      (const float*)d_in[7],   // b_g
      (const float*)d_in[8],   // W_kin
      (const float*)d_in[9],   // b_kin
      (const float*)d_in[10],  // W_kout
      (const float*)d_in[11],  // b_kout
      (float*)d_out);
}
